// VQVAE_78237124264402
// MI455X (gfx1250) — matrix operand order repaired and checked by reading, not compiled, not run on hardware
//
#include <hip/hip_runtime.h>
#include <hip/hip_bf16.h>

// ---------------------------------------------------------------------------
// VQ-VAE codebook update for MI455X (gfx1250, wave32, WMMA).
// N=131072 rows, D=64, K=1024 clusters.
// Distance GEMM: V_WMMA_F32_16X16X32_BF16 with f32 accumulate; codebook
// pre-packed into B-fragment layout and staged in LDS via async global->LDS.
// ---------------------------------------------------------------------------

typedef __attribute__((ext_vector_type(16))) __bf16 v16bf;
typedef __attribute__((ext_vector_type(8)))  float  v8f;

#define N_ROWS 131072
#define DDIM   64
#define KCLUST 1024
#define NTOT   8388608      // 32*64*64*64
#define NTILES (KCLUST / 16)
#define PACK_BYTES (KCLUST * DDIM * 2)          // 131072: packed bf16 codebook
#define C2P_BYTES  (NTILES * 32 * 4)            // 8192: per-lane ||c||^2 table
#define SMEM_BYTES (PACK_BYTES + C2P_BYTES)     // 139264

static __device__ __forceinline__ unsigned short f2bf_bits(float f) {
    unsigned u = __builtin_bit_cast(unsigned, f);
    return (unsigned short)((u + 0x7FFFu + ((u >> 16) & 1u)) >> 16);  // RNE
}
static __device__ __forceinline__ __bf16 bits2bf(unsigned short s) {
    return __builtin_bit_cast(__bf16, s);
}
static __device__ __forceinline__ void pack8(v16bf& a, int base, float4 p, float4 q) {
    a[base + 0] = bits2bf(f2bf_bits(p.x));
    a[base + 1] = bits2bf(f2bf_bits(p.y));
    a[base + 2] = bits2bf(f2bf_bits(p.z));
    a[base + 3] = bits2bf(f2bf_bits(p.w));
    a[base + 4] = bits2bf(f2bf_bits(q.x));
    a[base + 5] = bits2bf(f2bf_bits(q.y));
    a[base + 6] = bits2bf(f2bf_bits(q.z));
    a[base + 7] = bits2bf(f2bf_bits(q.w));
}

// ---------------------------------------------------------------------------
// Kernel 1: pack codebook into WMMA B-fragment layout, compute ||c_k||^2,
// zero atomic scratch.  pack[(ct*32 + lane)*32 + e]:
//   e 0..15  -> b0 element e : d = (e&8)*2 + hi*8 + (e&7)       (hi = lane>>4)
//   e 16..31 -> b1 element   : d = 32 + (e&8)*2 + hi*8 + (e&7)
//   column   = ct*16 + (lane&15)
// 65536 threads (256 blocks x 256).
// ---------------------------------------------------------------------------
__global__ void vq_prep(const float* __restrict__ codebook,
                        unsigned short* __restrict__ pack,
                        float* __restrict__ c2,
                        float* __restrict__ counts,
                        float* __restrict__ input_sum) {
    int t = blockIdx.x * 256 + threadIdx.x;      // 0 .. 65535
    int ct = t >> 10;                            // cluster tile
    int l  = (t >> 5) & 31;                      // lane
    int e  = t & 31;                             // fragment element
    int hi = l >> 4;
    int d  = ((e >> 4) << 5) + ((e & 8) << 1) + hi * 8 + (e & 7);
    int col = ct * 16 + (l & 15);
    pack[t] = f2bf_bits(codebook[d * KCLUST + col]);
    input_sum[t] = 0.0f;
    if (t < KCLUST) {
        counts[t] = 0.0f;
        float s = 0.0f;
        #pragma unroll 8
        for (int dd = 0; dd < DDIM; ++dd) {
            float v = codebook[dd * KCLUST + t];
            s += v * v;
        }
        c2[t] = s;
    }
}

// ---------------------------------------------------------------------------
// Kernel 2: nearest-codeword assignment via bf16 WMMA.
// Block = 128 threads (4 waves); each wave owns TWO 16-row strips (32 rows)
// and sweeps all 64 cluster tiles.  Packed codebook + c2 table staged in LDS
// once per block with async global->LDS (ASYNCcnt), then the inner loop is
// ds_load_b128 + v_wmma only.  Grid: 1024 blocks -> 4096 wave tasks.
// ---------------------------------------------------------------------------
__global__ void vq_assign(const float* __restrict__ x,
                          const unsigned short* __restrict__ pack,
                          const float* __restrict__ c2,
                          int* __restrict__ idx) {
    extern __shared__ char smem[];
    unsigned short* spack = (unsigned short*)smem;
    float*          sc2   = (float*)(smem + PACK_BYTES);

    const int tid = threadIdx.x;

    // ---- async stage: 128 threads x 1KB contiguous = 128KB ----------------
    {
        unsigned long long gbase = (unsigned long long)pack + (unsigned long long)tid * 1024ull;
        unsigned lbase = (unsigned)(unsigned long long)(smem) + (unsigned)tid * 1024u;
        #pragma unroll 8
        for (int i = 0; i < 64; ++i) {
            unsigned long long ga = gbase + (unsigned)(i * 16);
            unsigned la = lbase + (unsigned)(i * 16);
            asm volatile("global_load_async_to_lds_b128 %0, %1, off"
                         :: "v"(la), "v"(ga) : "memory");
        }
    }
    // ---- c2 table: sc2[ct*32 + lane] = ||c_{ct*16 + (lane&15)}||^2 --------
    for (int i = tid; i < NTILES * 32; i += 128)
        sc2[i] = c2[((i >> 5) << 4) + (i & 15)];
    asm volatile("s_wait_asynccnt 0" ::: "memory");
    __syncthreads();

    const int lane = tid & 31;
    const int wave = tid >> 5;
    const int task = blockIdx.x * 4 + wave;      // 0 .. 4095
    const int row0 = task * 32;                  // 32 rows per wave
    const int m  = lane & 15;
    const int hi = lane >> 4;

    // ---- A fragments: 2 strips x (16x32 bf16 pair covering D=64) ----------
    // 16-bit A layout: lanes 0-15 (M=lane) hold K={0..7,16..23};
    //                  lanes 16-31 (M=lane-16) hold K={8..15,24..31}.
    v16bf a0[2], a1[2];
    #pragma unroll
    for (int s = 0; s < 2; ++s) {
        const float4* xr4 = (const float4*)(x + (size_t)(row0 + s * 16 + m) * DDIM);
        float4 u0 = xr4[hi * 2 + 0],  u1 = xr4[hi * 2 + 1];   // K = c0..c0+7
        float4 u2 = xr4[hi * 2 + 4],  u3 = xr4[hi * 2 + 5];   // K = c0+16..c0+23
        float4 u4 = xr4[hi * 2 + 8],  u5 = xr4[hi * 2 + 9];   // +32
        float4 u6 = xr4[hi * 2 + 12], u7 = xr4[hi * 2 + 13];
        pack8(a0[s], 0, u0, u1);
        pack8(a0[s], 8, u2, u3);
        pack8(a1[s], 0, u4, u5);
        pack8(a1[s], 8, u6, u7);
    }

    // ---- running argmin: VGPR v holds row M = v + 8*hi, column N = m ------
    float bv[2][8];
    int   bi[2][8];
    #pragma unroll
    for (int s = 0; s < 2; ++s)
        #pragma unroll
        for (int v = 0; v < 8; ++v) { bv[s][v] = 3.4e38f; bi[s][v] = 0; }

    for (int ct = 0; ct < NTILES; ++ct) {
        const v16bf* bp = (const v16bf*)(spack + (size_t)(ct * 32 + lane) * 32);
        v16bf b0 = bp[0];
        v16bf b1 = bp[1];
        const float cc = sc2[ct * 32 + lane];
        const int col = ct * 16 + m;

        #pragma unroll
        for (int s = 0; s < 2; ++s) {
            v8f acc = {};
            acc = __builtin_amdgcn_wmma_f32_16x16x32_bf16(false, a0[s], false, b0,
                                                          (short)0, acc, false, false);
            acc = __builtin_amdgcn_wmma_f32_16x16x32_bf16(false, a1[s], false, b1,
                                                          (short)0, acc, false, false);
            #pragma unroll
            for (int v = 0; v < 8; ++v) {
                float sc = cc - 2.0f * acc[v];
                if (sc < bv[s][v]) { bv[s][v] = sc; bi[s][v] = col; }
            }
        }
    }

    // ---- argmin reduction across the 16 lanes of each half-wave -----------
    #pragma unroll
    for (int off = 8; off >= 1; off >>= 1) {
        #pragma unroll
        for (int s = 0; s < 2; ++s)
            #pragma unroll
            for (int v = 0; v < 8; ++v) {
                float ov = __shfl_xor(bv[s][v], off, 32);
                int   oi = __shfl_xor(bi[s][v], off, 32);
                if (ov < bv[s][v] || (ov == bv[s][v] && oi < bi[s][v])) {
                    bv[s][v] = ov; bi[s][v] = oi;
                }
            }
    }
    if (m == 0) {
        #pragma unroll
        for (int s = 0; s < 2; ++s)
            #pragma unroll
            for (int v = 0; v < 8; ++v)
                idx[row0 + s * 16 + v + 8 * hi] = bi[s][v];
    }
}

// ---------------------------------------------------------------------------
// Kernel 3: segment sums (counts + per-cluster input sums) via f32 atomics.
// ---------------------------------------------------------------------------
__global__ void vq_stats(const float* __restrict__ x,
                         const int* __restrict__ idx,
                         float* __restrict__ counts,
                         float* __restrict__ input_sum) {
    int t = blockIdx.x * 256 + threadIdx.x;
    int n = t >> 6;
    int d = t & 63;
    int k = idx[n];
    atomicAdd(&input_sum[d * KCLUST + k], x[t]);
    if (d == 0) atomicAdd(&counts[k], 1.0f);
}

// ---------------------------------------------------------------------------
// Kernel 4: EMA update + Laplace-normalized cluster sizes + new codebook.
// Single block of 1024 threads; LDS tree for the global sum.
// Also zeroes the loss slot for kernel 5's atomics.
// ---------------------------------------------------------------------------
__global__ void vq_ema(const float* __restrict__ ema_cs,
                       const float* __restrict__ ema_sum,
                       const float* __restrict__ counts,
                       const float* __restrict__ input_sum,
                       float* __restrict__ out_cs,
                       float* __restrict__ out_sum,
                       float* __restrict__ out_cb,
                       float* __restrict__ loss_slot) {
    __shared__ float red[KCLUST];
    const int k = threadIdx.x;
    const float decay = 0.99f, omd = 0.01f, eps = 1e-5f;

    float ncs = decay * ema_cs[k] + omd * counts[k];
    out_cs[k] = ncs;
    red[k] = ncs;
    __syncthreads();
    #pragma unroll
    for (int s = 512; s > 0; s >>= 1) {
        if (k < s) red[k] += red[k + s];
        __syncthreads();
    }
    const float n = red[0];
    const float cs = (ncs + eps) / (n + (float)KCLUST * eps) * n;
    const float inv = 1.0f / cs;

    for (int d = 0; d < DDIM; ++d) {
        float ns = decay * ema_sum[d * KCLUST + k] + omd * input_sum[d * KCLUST + k];
        out_sum[d * KCLUST + k] = ns;
        out_cb[d * KCLUST + k]  = ns * inv;
    }
    if (k == 0) *loss_slot = 0.0f;
}

// ---------------------------------------------------------------------------
// Kernel 5: quantize with the UPDATED codebook + embedding loss.
// quantized_st == quantized numerically (x + sg(q - x) = q in fwd).
// ---------------------------------------------------------------------------
__global__ void vq_quant(const float* __restrict__ x,
                         const int* __restrict__ idx,
                         const float* __restrict__ new_cb,
                         float* __restrict__ out_q,
                         float* __restrict__ loss_slot) {
    __shared__ float red[256];
    int t = blockIdx.x * 256 + threadIdx.x;
    int n = t >> 6;
    int d = t & 63;
    int k = idx[n];
    float q = new_cb[d * KCLUST + k];
    out_q[t] = q;
    float diff = q - x[t];
    red[threadIdx.x] = diff * diff;
    __syncthreads();
    #pragma unroll
    for (int s = 128; s > 0; s >>= 1) {
        if (threadIdx.x < s) red[threadIdx.x] += red[threadIdx.x + s];
        __syncthreads();
    }
    if (threadIdx.x == 0)
        atomicAdd(loss_slot, red[0] * (1.0f / (float)NTOT));
}

// ---------------------------------------------------------------------------
extern "C" void kernel_launch(void* const* d_in, const int* in_sizes, int n_in,
                              void* d_out, int out_size, void* d_ws, size_t ws_size,
                              hipStream_t stream) {
    const float* inputs   = (const float*)d_in[0];   // [32,64,64,64]
    const float* codebook = (const float*)d_in[1];   // [64,1024]
    const float* ema_cs   = (const float*)d_in[2];   // [1024]
    const float* ema_sum  = (const float*)d_in[3];   // [64,1024]

    float* out = (float*)d_out;
    float* out_q    = out;                           // [8388608]
    float* out_loss = out + NTOT;                    // [1]
    float* out_cb   = out + NTOT + 1;                // [65536]
    float* out_cs   = out + NTOT + 1 + DDIM * KCLUST;          // [1024]
    float* out_sum  = out + NTOT + 1 + DDIM * KCLUST + KCLUST; // [65536]

    char* ws = (char*)d_ws;
    int*            idx       = (int*)(ws + 0);                 // 524288 B
    float*          counts    = (float*)(ws + 524288);          //   4096 B
    float*          input_sum = (float*)(ws + 528384);          // 262144 B
    float*          c2        = (float*)(ws + 790528);          //   4096 B
    unsigned short* pack      = (unsigned short*)(ws + 794624); // 131072 B

    vq_prep  <<<256,   256, 0,          stream>>>(codebook, pack, c2, counts, input_sum);
    vq_assign<<<1024,  128, SMEM_BYTES, stream>>>(inputs, pack, c2, idx);
    vq_stats <<<32768, 256, 0,          stream>>>(inputs, idx, counts, input_sum);
    vq_ema   <<<1,    1024, 0,          stream>>>(ema_cs, ema_sum, counts, input_sum,
                                                  out_cs, out_sum, out_cb, out_loss);
    vq_quant <<<32768, 256, 0,          stream>>>(inputs, idx, out_cb, out_q, out_loss);
}
